// MessageLayer_14078902796472
// MI455X (gfx1250) — matrix operand order repaired and checked by reading, not compiled, not run on hardware
//
#include <hip/hip_runtime.h>
#include <hip/hip_bf16.h>
#include <cstdint>

typedef __attribute__((ext_vector_type(16))) _Float16 v16h;
typedef __attribute__((ext_vector_type(8)))  _Float16 v8h;
typedef __attribute__((ext_vector_type(8)))  float    v8f;

#define HEADS 3
#define ELEMF 64
#define EDGEF 128
#define HIDF  256
#define CATF  256
#define WAVES 8
#define ROWS_PER_BLOCK (WAVES * 16)   // 128 edges per block
#define THREADS (WAVES * 32)          // 256 threads = 8 wave32

__device__ __forceinline__ float lrelu(float x) { return x > 0.f ? x : 0.01f * x; }

// ---- WMMA fragment helpers (V_WMMA_F32_16X16X32_F16, wave32) ----
// A fragment: 16x32 f16 tile from row-major storage (stride in halves).
// ISA layout: lane L holds row (L&15); halves 0..7 -> K = kbase+(L<16?0:8)+i,
//             halves 8..15 -> K = kbase+16+(L<16?0:8)+i
__device__ __forceinline__ v16h a_frag(const _Float16* base, int strideH, int kbase, int lane) {
  const _Float16* r = base + (size_t)(lane & 15) * strideH + kbase + ((lane & 16) ? 8 : 0);
  v8h lo = *(const v8h*)(r);
  v8h hi = *(const v8h*)(r + 16);
  return __builtin_shufflevector(lo, hi, 0,1,2,3,4,5,6,7,8,9,10,11,12,13,14,15);
}

// B fragment from pre-packed weights: each 32(K)x16(N) tile stored so a lane's
// 16 halves are contiguous (32B aligned): idx = ((kt*ntT+nt)*32+lane)*16+elem,
// K = kt*32 + (lane<16?0:8) + (elem&7) + (elem<8?0:16), N = nt*16 + (lane&15)
__device__ __forceinline__ v16h b_frag(const _Float16* packed, int ntTiles, int kt, int nt, int lane) {
  return *(const v16h*)(packed + ((((size_t)kt * ntTiles + nt) * 32 + lane) << 4));
}

#define WMMA(a, b, c) __builtin_amdgcn_wmma_f32_16x16x32_f16(false, (a), false, (b), (short)0, (c), false, false)

// ---- CDNA5 async global->LDS staging (GLOBAL_LOAD_ASYNC_TO_LDS_B128, ASYNCcnt) ----
// Each thread copies strided 16B chunks; whole block must barrier after WAIT_ASYNC.
__device__ __forceinline__ void async_stage(const _Float16* __restrict__ g, _Float16* lds, int halves) {
  for (int i = threadIdx.x; i < (halves >> 3); i += THREADS) {
    unsigned dst = (unsigned)(size_t)(lds + (size_t)i * 8);             // LDS byte address
    unsigned long long src = (unsigned long long)(size_t)(g + (size_t)i * 8);
    asm volatile("global_load_async_to_lds_b128 %0, %1, off" :: "v"(dst), "v"(src) : "memory");
  }
}
#define WAIT_ASYNC() asm volatile("s_wait_asynccnt 0x0" ::: "memory")

__device__ __forceinline__ void atomicMaxF(float* addr, float v) {
  if (v >= 0.f) atomicMax((int*)addr, __float_as_int(v));
  else          atomicMin((unsigned int*)addr, (unsigned int)__float_as_int(v));
}

// ---- repack a KxN fp32 weight matrix into f16 WMMA-B tile order ----
__global__ void k_pack_b(const float* __restrict__ W, _Float16* __restrict__ out, int Ktot, int Ntot) {
  int tid = blockIdx.x * blockDim.x + threadIdx.x;
  if (tid >= Ktot * Ntot) return;
  int elem = tid & 15;
  int lane = (tid >> 4) & 31;
  int tile = tid >> 9;
  int ntT = Ntot >> 4;
  int kt = tile / ntT, nt = tile - kt * ntT;
  int k = kt * 32 + ((lane & 16) ? 8 : 0) + (elem & 7) + ((elem & 8) ? 16 : 0);
  int n = nt * 16 + (lane & 15);
  out[tid] = (_Float16)W[(size_t)k * Ntot + n];
}

__global__ void k_f32_to_f16(const float* __restrict__ in, _Float16* __restrict__ out, long n) {
  long t = (long)blockIdx.x * blockDim.x + threadIdx.x;
  if (t < n) out[t] = (_Float16)in[t];
}

// fea[:,0:64]=elem_in[self], fea[:,64:128]=elem_in[nbr]  (f16, row stride CATF)
__global__ void k_gather(const float* __restrict__ elem_in, const int* __restrict__ selfI,
                         const int* __restrict__ nbrI, _Float16* __restrict__ feaH, long M) {
  long t = (long)blockIdx.x * blockDim.x + threadIdx.x;
  long e = t >> 7;
  if (e >= M) return;
  int c = (int)(t & 127);
  int node = (c < ELEMF) ? selfI[e] : nbrI[e];
  feaH[e * CATF + c] = (_Float16)elem_in[(size_t)node * ELEMF + (c & (ELEMF - 1))];
}

struct EdgeBias { const float* b[5]; };

// Fused 5-layer edge MLP (4x [128->128]+LeakyReLU, then 128->128 linear).
// All 5 weight matrices async-staged to LDS once per block; each wave owns
// 16 edges with activations ping-ponging in wave-private LDS.
__global__ void __launch_bounds__(THREADS) k_edge_mlp(
    const _Float16* __restrict__ edgeH, const _Float16* __restrict__ wpack,
    EdgeBias eb, _Float16* __restrict__ feaH, long M) {
  __shared__ __align__(16) _Float16 wls[5 * EDGEF * EDGEF];   // 160 KB
  __shared__ __align__(16) _Float16 act[WAVES][2][16 * 136];  // 68 KB
  async_stage(wpack, wls, 5 * EDGEF * EDGEF);
  WAIT_ASYNC();
  __syncthreads();

  int lane = threadIdx.x & 31;
  int wave = threadIdx.x >> 5;
  long edge0 = (long)blockIdx.x * ROWS_PER_BLOCK + wave * 16;
  if (edge0 >= M) return;   // after the staging barrier: safe
  int col = lane & 15;
  int rhi = (lane & 16) ? 8 : 0;
  #pragma unroll
  for (int l = 0; l < 5; ++l) {
    const _Float16* w = wls + l * (EDGEF * EDGEF);
    const _Float16* src;
    int srcStride;
    if (l == 0) { src = edgeH + edge0 * EDGEF; srcStride = EDGEF; }
    else        { src = act[wave][(l - 1) & 1]; srcStride = 136; }
    _Float16* dst = act[wave][l & 1];
    #pragma unroll
    for (int nt = 0; nt < EDGEF / 16; ++nt) {
      v8f acc = {};
      #pragma unroll
      for (int kt = 0; kt < EDGEF / 32; ++kt)
        acc = WMMA(a_frag(src, srcStride, kt * 32, lane), b_frag(w, EDGEF / 16, kt, nt, lane), acc);
      float bias = eb.b[l][nt * 16 + col];
      if (l < 4) {
        #pragma unroll
        for (int j = 0; j < 8; ++j)
          dst[(size_t)(rhi + j) * 136 + nt * 16 + col] = (_Float16)lrelu(acc[j] + bias);
      } else {
        #pragma unroll
        for (int j = 0; j < 8; ++j)
          feaH[(edge0 + rhi + j) * CATF + EDGEF + nt * 16 + col] = (_Float16)(acc[j] + bias);
      }
    }
    asm volatile("s_wait_dscnt 0" ::: "memory");  // wave-private LDS ping-pong
  }
}

// gate head: hid = LeakyReLU(fea @ W1 + b1) via WMMA (W1 async-staged in LDS);
// logit = hid . W2 + b2
__global__ void __launch_bounds__(THREADS) k_gate(
    const _Float16* __restrict__ feaH, const _Float16* __restrict__ w1pack,
    const float* __restrict__ b1, const float* __restrict__ w2,
    const float* __restrict__ b2, float* __restrict__ glogit, long M) {
  __shared__ __align__(16) _Float16 w1s[CATF * HIDF];       // 128 KB
  __shared__ __align__(16) _Float16 hid[WAVES][16 * 264];   // 66 KB
  int h = blockIdx.y;
  async_stage(w1pack + (size_t)h * (CATF * HIDF), w1s, CATF * HIDF);
  WAIT_ASYNC();
  __syncthreads();

  int lane = threadIdx.x & 31;
  int wave = threadIdx.x >> 5;
  long edge0 = (long)blockIdx.x * ROWS_PER_BLOCK + wave * 16;
  if (edge0 >= M) return;
  const _Float16* A = feaH + edge0 * CATF;
  v16h af[CATF / 32];
  #pragma unroll
  for (int kt = 0; kt < CATF / 32; ++kt) af[kt] = a_frag(A, CATF, kt * 32, lane);
  int col = lane & 15;
  int rhi = (lane & 16) ? 8 : 0;
  #pragma unroll
  for (int nt = 0; nt < HIDF / 16; ++nt) {
    v8f acc = {};
    #pragma unroll
    for (int kt = 0; kt < CATF / 32; ++kt)
      acc = WMMA(af[kt], b_frag(w1s, HIDF / 16, kt, nt, lane), acc);
    float bias = b1[h * HIDF + nt * 16 + col];
    #pragma unroll
    for (int j = 0; j < 8; ++j)
      hid[wave][(size_t)(rhi + j) * 264 + nt * 16 + col] = (_Float16)lrelu(acc[j] + bias);
  }
  asm volatile("s_wait_dscnt 0" ::: "memory");
  if (lane < 16) {
    const _Float16* hr = hid[wave] + (size_t)lane * 264;
    const float* wv = w2 + (size_t)h * HIDF;
    float s = b2[h];
    for (int k = 0; k < HIDF; ++k) s += (float)hr[k] * wv[k];
    glogit[(size_t)h * M + edge0 + lane] = s;
  }
}

// msg head: m = (LeakyReLU(fea@W1+b1))@W2+b2 via WMMA (W1,W2 async-staged);
// scatter g*m into headAcc with f32 global atomics
__global__ void __launch_bounds__(THREADS) k_msg(
    const _Float16* __restrict__ feaH, const _Float16* __restrict__ w1pack,
    const float* __restrict__ b1, const _Float16* __restrict__ w2pack,
    const float* __restrict__ b2, const float* __restrict__ gbuf,
    const int* __restrict__ selfI, float* __restrict__ headAcc, long M) {
  __shared__ __align__(16) _Float16 w1s[CATF * HIDF];       // 128 KB
  __shared__ __align__(16) _Float16 w2s[HIDF * ELEMF];      // 32 KB
  __shared__ __align__(16) _Float16 hid[WAVES][16 * 264];   // 66 KB
  __shared__ float gsh[WAVES][16];
  __shared__ int   nsh[WAVES][16];
  int h = blockIdx.y;
  async_stage(w1pack + (size_t)h * (CATF * HIDF), w1s, CATF * HIDF);
  async_stage(w2pack + (size_t)h * (HIDF * ELEMF), w2s, HIDF * ELEMF);
  WAIT_ASYNC();
  __syncthreads();

  int lane = threadIdx.x & 31;
  int wave = threadIdx.x >> 5;
  long edge0 = (long)blockIdx.x * ROWS_PER_BLOCK + wave * 16;
  if (edge0 >= M) return;
  if (lane < 16) {
    gsh[wave][lane] = gbuf[(size_t)h * M + edge0 + lane];
    nsh[wave][lane] = selfI[edge0 + lane];
  }
  const _Float16* A = feaH + edge0 * CATF;
  v16h af[CATF / 32];
  #pragma unroll
  for (int kt = 0; kt < CATF / 32; ++kt) af[kt] = a_frag(A, CATF, kt * 32, lane);
  int col = lane & 15;
  int rhi = (lane & 16) ? 8 : 0;
  #pragma unroll
  for (int nt = 0; nt < HIDF / 16; ++nt) {
    v8f acc = {};
    #pragma unroll
    for (int kt = 0; kt < CATF / 32; ++kt)
      acc = WMMA(af[kt], b_frag(w1s, HIDF / 16, kt, nt, lane), acc);
    float bias = b1[h * HIDF + nt * 16 + col];
    #pragma unroll
    for (int j = 0; j < 8; ++j)
      hid[wave][(size_t)(rhi + j) * 264 + nt * 16 + col] = (_Float16)lrelu(acc[j] + bias);
  }
  asm volatile("s_wait_dscnt 0" ::: "memory");
  #pragma unroll
  for (int nt = 0; nt < ELEMF / 16; ++nt) {
    v8f acc = {};
    #pragma unroll
    for (int kt = 0; kt < HIDF / 32; ++kt)
      acc = WMMA(a_frag(hid[wave], 264, kt * 32, lane), b_frag(w2s, ELEMF / 16, kt, nt, lane), acc);
    float bias = b2[h * ELEMF + nt * 16 + col];
    #pragma unroll
    for (int j = 0; j < 8; ++j) {
      int r = rhi + j;
      float v = (acc[j] + bias) * gsh[wave][r];
      unsafeAtomicAdd(headAcc + (size_t)nsh[wave][r] * ELEMF + nt * 16 + col, v);
    }
  }
}

__global__ void k_init(float* smax, float* ssum, float* headAcc, long n3, long nAcc) {
  long t = (long)blockIdx.x * blockDim.x + threadIdx.x;
  if (t < n3) { smax[t] = -__builtin_inff(); ssum[t] = 0.f; }
  if (t < nAcc) headAcc[t] = 0.f;
}

__global__ void k_segmax(const float* __restrict__ glogit, const int* __restrict__ selfI,
                         float* __restrict__ smax, long M, int N) {
  long t = (long)blockIdx.x * blockDim.x + threadIdx.x;
  if (t >= M) return;
  int s = selfI[t];
  #pragma unroll
  for (int h = 0; h < HEADS; ++h) atomicMaxF(smax + (size_t)h * N + s, glogit[(size_t)h * M + t]);
}

__global__ void k_exp(const float* __restrict__ glogit, const float* __restrict__ smax,
                      const float* __restrict__ wts, const int* __restrict__ selfI,
                      const int* __restrict__ nbrI, const float* __restrict__ pow_p,
                      float* __restrict__ gbuf, float* __restrict__ ssum, long M, int N) {
  long t = (long)blockIdx.x * blockDim.x + threadIdx.x;
  if (t >= M) return;
  int s = selfI[t];
  float w = wts[nbrI[t]];
  #pragma unroll
  for (int h = 0; h < HEADS; ++h) {
    float g = __powf(w, pow_p[h]) * __expf(glogit[(size_t)h * M + t] - smax[(size_t)h * N + s]);
    gbuf[(size_t)h * M + t] = g;
    unsafeAtomicAdd(ssum + (size_t)h * N + s, g);
  }
}

__global__ void k_norm(const float* __restrict__ ssum, const int* __restrict__ selfI,
                       float* __restrict__ gbuf, float* __restrict__ gatesOut, long M, int N) {
  long t = (long)blockIdx.x * blockDim.x + threadIdx.x;
  if (t >= M) return;
  int s = selfI[t];
  #pragma unroll
  for (int h = 0; h < HEADS; ++h) {
    float g = gbuf[(size_t)h * M + t] / (ssum[(size_t)h * N + s] + 1e-10f);
    gbuf[(size_t)h * M + t] = g;
    gatesOut[(size_t)h * M + t] = g;
  }
}

__global__ void k_final(const float* __restrict__ headAcc, const float* __restrict__ elem_in,
                        float* __restrict__ out, long n) {
  long t = (long)blockIdx.x * blockDim.x + threadIdx.x;
  if (t < n) out[t] = headAcc[t] * (1.f / 3.f) + elem_in[t];
}

extern "C" void kernel_launch(void* const* d_in, const int* in_sizes, int n_in,
                              void* d_out, int out_size, void* d_ws, size_t ws_size,
                              hipStream_t stream) {
  const float* elem_w   = (const float*)d_in[0];
  const float* elem_in  = (const float*)d_in[1];
  const float* edge_fea = (const float*)d_in[2];
  const int*   selfI    = (const int*)d_in[3];
  const int*   nbrI     = (const int*)d_in[4];
  const float* edgeW[5] = {(const float*)d_in[5], (const float*)d_in[7], (const float*)d_in[9],
                           (const float*)d_in[11], (const float*)d_in[13]};
  EdgeBias eb = {{(const float*)d_in[6], (const float*)d_in[8], (const float*)d_in[10],
                  (const float*)d_in[12], (const float*)d_in[14]}};
  const float* gateW1 = (const float*)d_in[15];
  const float* gateB1 = (const float*)d_in[16];
  const float* gateW2 = (const float*)d_in[17];
  const float* gateB2 = (const float*)d_in[18];
  const float* msgW1  = (const float*)d_in[19];
  const float* msgB1  = (const float*)d_in[20];
  const float* msgW2  = (const float*)d_in[21];
  const float* msgB2  = (const float*)d_in[22];
  const float* powP   = (const float*)d_in[23];

  const long N = in_sizes[0];   // elem_weights is (N,1)
  const long M = in_sizes[3];   // self_fea_idx is (M,)

  char* ws = (char*)d_ws;
  size_t off = 0;
  auto take = [&](size_t bytes) -> void* {
    void* p = ws + off;
    off = (off + bytes + 255) & ~(size_t)255;
    return p;
  };
  _Float16* edgeH  = (_Float16*)take((size_t)M * EDGEF * 2);
  _Float16* feaH   = (_Float16*)take((size_t)M * CATF * 2);
  _Float16* packE  = (_Float16*)take((size_t)5 * EDGEF * EDGEF * 2);
  _Float16* packG1 = (_Float16*)take((size_t)HEADS * CATF * HIDF * 2);
  _Float16* packM1 = (_Float16*)take((size_t)HEADS * CATF * HIDF * 2);
  _Float16* packM2 = (_Float16*)take((size_t)HEADS * HIDF * ELEMF * 2);
  float* glogit  = (float*)take((size_t)HEADS * M * 4);
  float* gbuf    = (float*)take((size_t)HEADS * M * 4);
  float* smax    = (float*)take((size_t)HEADS * N * 4);
  float* ssum    = (float*)take((size_t)HEADS * N * 4);
  float* headAcc = (float*)take((size_t)N * ELEMF * 4);
  (void)ws_size; (void)n_in; (void)out_size;

  float* out = (float*)d_out;
  float* gatesOut = out + (size_t)N * ELEMF;

  // --- weight repack (tiny, L2 resident afterwards) ---
  for (int l = 0; l < 5; ++l)
    k_pack_b<<<(EDGEF * EDGEF + 255) / 256, 256, 0, stream>>>(edgeW[l], packE + (size_t)l * EDGEF * EDGEF, EDGEF, EDGEF);
  for (int h = 0; h < HEADS; ++h) {
    k_pack_b<<<(CATF * HIDF + 255) / 256, 256, 0, stream>>>(gateW1 + (size_t)h * CATF * HIDF, packG1 + (size_t)h * CATF * HIDF, CATF, HIDF);
    k_pack_b<<<(CATF * HIDF + 255) / 256, 256, 0, stream>>>(msgW1 + (size_t)h * CATF * HIDF, packM1 + (size_t)h * CATF * HIDF, CATF, HIDF);
    k_pack_b<<<(HIDF * ELEMF + 255) / 256, 256, 0, stream>>>(msgW2 + (size_t)h * HIDF * ELEMF, packM2 + (size_t)h * HIDF * ELEMF, HIDF, ELEMF);
  }

  long nconv = M * EDGEF;
  k_f32_to_f16<<<(unsigned)((nconv + 255) / 256), 256, 0, stream>>>(edge_fea, edgeH, nconv);
  k_gather<<<(unsigned)((M * 128 + 255) / 256), 256, 0, stream>>>(elem_in, selfI, nbrI, feaH, M);

  unsigned nblk = (unsigned)((M + ROWS_PER_BLOCK - 1) / ROWS_PER_BLOCK);
  k_edge_mlp<<<nblk, THREADS, 0, stream>>>(edgeH, packE, eb, feaH, M);

  long nAcc = N * ELEMF, n3 = (long)HEADS * N;
  long ninit = nAcc > n3 ? nAcc : n3;
  k_init<<<(unsigned)((ninit + 255) / 256), 256, 0, stream>>>(smax, ssum, headAcc, n3, nAcc);

  k_gate<<<dim3(nblk, HEADS), THREADS, 0, stream>>>(feaH, packG1, gateB1, gateW2, gateB2, glogit, M);
  k_segmax<<<(unsigned)((M + 255) / 256), 256, 0, stream>>>(glogit, selfI, smax, M, (int)N);
  k_exp<<<(unsigned)((M + 255) / 256), 256, 0, stream>>>(glogit, smax, elem_w, selfI, nbrI, powP, gbuf, ssum, M, (int)N);
  k_norm<<<(unsigned)((M + 255) / 256), 256, 0, stream>>>(ssum, selfI, gbuf, gatesOut, M, (int)N);
  k_msg<<<dim3(nblk, HEADS), THREADS, 0, stream>>>(feaH, packM1, msgB1, packM2, msgB2, gbuf, selfI, headAcc, M);
  k_final<<<(unsigned)((nAcc + 255) / 256), 256, 0, stream>>>(headAcc, elem_in, out, nAcc);
}
